// NCELoss_75187697484235
// MI455X (gfx1250) — matrix operand order repaired and checked by reading, not compiled, not run on hardware
//
#include <hip/hip_runtime.h>

// ---- problem constants (from reference) ----
#define VOCAB      100000
#define DIMD       256
#define BATCH      32
#define SEQN       128
#define KNOISE     100
#define NORM_TERM  11.512925f
#define LOG_K      4.60517019f          // log(100)
#define TOKENS     (BATCH * SEQN)       // 4096
#define WPB        8                    // waves per block (wave32)
#define NSLOT      112                  // 7 tiles of 16 (1 target + 100 noise + 11 pad)
#define NTILE      (NSLOT / 16)

typedef __attribute__((ext_vector_type(16))) _Float16 v16h;
typedef __attribute__((ext_vector_type(8)))  _Float16 v8h;
typedef __attribute__((ext_vector_type(8)))  float    v8f;
typedef __attribute__((ext_vector_type(4)))  float    vf4;

__device__ __forceinline__ float softplusf(float x) {
    // softplus(x) = log(1+exp(x)); guard large x
    float r = __logf(1.0f + __expf(x));
    return (x > 15.0f) ? x : r;
}

__global__ void nce_zero_kernel(float* out) { out[0] = 0.0f; }

__global__ __launch_bounds__(WPB * 32)
void nce_loss_wmma_kernel(const int*   __restrict__ target,       // (B,N)
                          const int*   __restrict__ noise,        // (B,N,K)
                          const float* __restrict__ hidden,       // (B,N,D)
                          const float* __restrict__ weight,       // (V,D)
                          const float* __restrict__ bias,         // (V,)
                          const float* __restrict__ nprobs,       // (V,)
                          float*       __restrict__ out)          // scalar
{
    __shared__ __align__(16) _Float16 hs[WPB][DIMD];  // f16 copy of hidden per wave

    const int lane  = threadIdx.x & 31;               // wave32 (gfx1250)
    const int wv    = threadIdx.x >> 5;
    const int j     = lane & 15;                      // candidate column within tile
    const int half  = lane >> 4;                      // lane half (0/1)
    const int token = blockIdx.x * WPB + wv;          // 0..4095

    // ---- stage this token's hidden vector into LDS as f16 ----
    const float* hrow = hidden + (size_t)token * DIMD;
    {
        vf4 h0 = ((const vf4*)hrow)[lane * 2 + 0];
        vf4 h1 = ((const vf4*)hrow)[lane * 2 + 1];
        v8h hh;
        #pragma unroll
        for (int i = 0; i < 4; ++i) { hh[i] = (_Float16)h0[i]; hh[4 + i] = (_Float16)h1[i]; }
        *(v8h*)&hs[wv][lane * 8] = hh;
    }
    __syncthreads();

    // ---- build the 8 replicated-A fragments (A 16x32 f16, all M rows == h chunk) ----
    // lane-half h holds contiguous K = d0+16h .. d0+16h+15 (matches B)
    v16h afrag[8];
    #pragma unroll
    for (int s = 0; s < 8; ++s) {
        const int d0 = s * 32;
        v8h lo = *(const v8h*)&hs[wv][d0 + half * 16];
        v8h hi = *(const v8h*)&hs[wv][d0 + half * 16 + 8];
        v16h a;
        #pragma unroll
        for (int i = 0; i < 8; ++i) { a[i] = lo[i]; a[8 + i] = hi[i]; }
        afrag[s] = a;
    }

    const int tgt = target[token];                    // wave-uniform scalar gather
    const int* nrow = noise + (size_t)token * KNOISE;

    float acc = 0.0f;

    #pragma unroll 1
    for (int t = 0; t < NTILE; ++t) {
        const int slot = t * 16 + j;
        const bool isTgt   = (slot == 0);
        const bool isNoise = (slot >= 1) && (slot <= KNOISE);
        int nIdx = slot - 1;
        nIdx = nIdx < 0 ? 0 : (nIdx > KNOISE - 1 ? KNOISE - 1 : nIdx);
        const int cand = isTgt ? tgt : nrow[nIdx];    // clamped (safe) for pad slots

        const float* wrow = weight + (size_t)cand * DIMD;
        __builtin_prefetch(wrow, 0, 3);               // -> global_prefetch_b8

        v8f c = {};
        #pragma unroll
        for (int s = 0; s < 8; ++s) {
            const int d0 = s * 32;
            // B 32x16 f16: lane-half h holds contiguous K = d0+16h .. d0+16h+15
            const vf4* r = (const vf4*)(wrow + d0 + half * 16);
            vf4 q0 = r[0], q1 = r[1], q2 = r[2], q3 = r[3];
            v16h bf;
            #pragma unroll
            for (int i = 0; i < 4; ++i) {
                bf[i]      = (_Float16)q0[i];
                bf[4 + i]  = (_Float16)q1[i];
                bf[8 + i]  = (_Float16)q2[i];
                bf[12 + i] = (_Float16)q3[i];
            }
            // D = A x B + C  (16x16x32, f32 accumulate)
            c = __builtin_amdgcn_wmma_f32_16x16x32_f16(
                    false, afrag[s], false, bf, (short)0, c, false, false);
        }

        // c[0]: lanes 0-15 -> (M=0, N=j); lanes 16-31 -> (M=8, N=j). Rows identical;
        // keep half 0 only. Branchless to keep EXEC all-ones around WMMA.
        const float dotv  = c[0];
        const float score = dotv + bias[cand] - NORM_TERM;
        const float termT = softplusf(LOG_K - score);                      // label=1
        const float sn    = __logf(nprobs[cand]);                          // log P_noise
        const float termN = softplusf(score - sn - LOG_K);                 // label=0
        float term = isTgt ? termT : (isNoise ? termN : 0.0f);
        term = (half == 0) ? term : 0.0f;
        acc += term;
    }

    // ---- wave reduction, then global accumulate of the mean ----
    #pragma unroll
    for (int off = 16; off >= 1; off >>= 1)
        acc += __shfl_xor(acc, off, 32);
    if (lane == 0)
        atomicAdd(out, acc * (1.0f / (float)TOKENS));
}

extern "C" void kernel_launch(void* const* d_in, const int* in_sizes, int n_in,
                              void* d_out, int out_size, void* d_ws, size_t ws_size,
                              hipStream_t stream) {
    (void)in_sizes; (void)n_in; (void)out_size; (void)d_ws; (void)ws_size;
    const int*   target = (const int*)  d_in[0];
    const int*   noise  = (const int*)  d_in[1];
    const float* hidden = (const float*)d_in[2];
    const float* weight = (const float*)d_in[3];
    const float* bias   = (const float*)d_in[4];
    const float* nprobs = (const float*)d_in[5];
    float* out = (float*)d_out;

    nce_zero_kernel<<<1, 1, 0, stream>>>(out);
    nce_loss_wmma_kernel<<<dim3(TOKENS / WPB), dim3(WPB * 32), 0, stream>>>(
        target, noise, hidden, weight, bias, nprobs, out);
}
